// SolarGateReference_10840497455877
// MI455X (gfx1250) — matrix operand order repaired and checked by reading, not compiled, not run on hardware
//
#include <hip/hip_runtime.h>
#include <hip/hip_bf16.h>

// MoE sigmoid-gating router for MI455X (gfx1250, wave32, WMMA).
//   logits[t][e] = sum_k x[t][k] * W[e][k]     (bf16 WMMA, f32 accum)
//   scores = sigmoid(logits); top-8 by scores+bias; weights = 2.5*s/sum(s)
//
// Kernel 1: convert W (256x7168 f32, L2-resident) to bf16 once into d_ws.
// Kernel 2: block = 128 threads (4 waves) owns 32 tokens; each wave computes
//           a 32x64 logit tile (2 M-subtiles x 4 N-tiles = 8 WMMAs / K-step).
//           Grid = tokens/32 = 512 blocks; all 256 experts covered per block.

#define HIDDEN          7168
#define N_EXPERTS       256
#define TOPK            8
#define M_TILE          32
#define ROUTED_SCALING  2.5f

typedef __attribute__((ext_vector_type(16))) __bf16 v16bf;
typedef __attribute__((ext_vector_type(8)))  __bf16 v8bf;
typedef __attribute__((ext_vector_type(8)))  float  v8f;

// Load 8+8 consecutive f32 and pack to a 16-element bf16 fragment.
__device__ __forceinline__ v16bf load_cvt_frag(const float* __restrict__ lo,
                                               const float* __restrict__ hi) {
  float4 a = *(const float4*)(lo);
  float4 b = *(const float4*)(lo + 4);
  float4 c = *(const float4*)(hi);
  float4 d = *(const float4*)(hi + 4);
  v16bf r;
  r[0]  = (__bf16)a.x; r[1]  = (__bf16)a.y; r[2]  = (__bf16)a.z; r[3]  = (__bf16)a.w;
  r[4]  = (__bf16)b.x; r[5]  = (__bf16)b.y; r[6]  = (__bf16)b.z; r[7]  = (__bf16)b.w;
  r[8]  = (__bf16)c.x; r[9]  = (__bf16)c.y; r[10] = (__bf16)c.z; r[11] = (__bf16)c.w;
  r[12] = (__bf16)d.x; r[13] = (__bf16)d.y; r[14] = (__bf16)d.z; r[15] = (__bf16)d.w;
  return r;
}

// ---- Kernel 1: f32 -> bf16 weight conversion (runs once per launch) ----
__global__ __launch_bounds__(256)
void convert_w_bf16_kernel(const float* __restrict__ w, __bf16* __restrict__ wb, int n) {
  int i = (blockIdx.x * blockDim.x + threadIdx.x) * 8;
  if (i + 8 <= n) {
    float4 a = *(const float4*)(w + i);
    float4 b = *(const float4*)(w + i + 4);
    v8bf r;
    r[0] = (__bf16)a.x; r[1] = (__bf16)a.y; r[2] = (__bf16)a.z; r[3] = (__bf16)a.w;
    r[4] = (__bf16)b.x; r[5] = (__bf16)b.y; r[6] = (__bf16)b.z; r[7] = (__bf16)b.w;
    *(v8bf*)(wb + i) = r;
  }
}

// ---- Kernel 2: GEMM + sigmoid + top-8 ----
template <bool PRECONV>
__global__ __launch_bounds__(128)
void moe_gate_topk_kernel(const float*  __restrict__ x,
                          const float*  __restrict__ wf,
                          const __bf16* __restrict__ wb,
                          const float*  __restrict__ bias,
                          int*   __restrict__ out_idx,
                          float* __restrict__ out_w)
{
  __shared__ float logits_lds[M_TILE][N_EXPERTS];   // 32 KB

  const int token0 = blockIdx.x * M_TILE;
  const int lane   = threadIdx.x & 31;
  const int wave   = threadIdx.x >> 5;              // 0..3
  const int l15    = lane & 15;
  const int lhalf  = lane >> 4;                     // 0 or 1

  const int e_w = wave * 64;                        // 4 N-tiles of 16 experts

  const float* xrow0 = x + (size_t)(token0 + l15) * HIDDEN;       // M-subtile 0
  const float* xrow1 = xrow0 + (size_t)16 * HIDDEN;               // M-subtile 1

  const __bf16* wbrow[4];
  const float*  wfrow[4];
  #pragma unroll
  for (int j = 0; j < 4; ++j) {
    const int e = e_w + j * 16 + l15;               // B column (expert)
    wbrow[j] = wb + (size_t)e * HIDDEN;
    wfrow[j] = wf + (size_t)e * HIDDEN;
  }

  v8f acc[2][4];
  #pragma unroll
  for (int m = 0; m < 2; ++m)
    #pragma unroll
    for (int j = 0; j < 4; ++j)
      acc[m][j] = (v8f){};

  for (int k0 = 0; k0 < HIDDEN; k0 += 32) {
    // A (16-bit 16x32): lanes 0-15: elems 0-15 = K+0..15
    //                   lanes16-31: elems 0-15 = K+16..31
    const int ka = k0 + lhalf * 16;
    v16bf a0 = load_cvt_frag(xrow0 + ka, xrow0 + ka + 8);
    v16bf a1 = load_cvt_frag(xrow1 + ka, xrow1 + ka + 8);

    // B (16-bit 32x16): lanes 0-15: elems 0-15 = K+0..15 (col = e_base+lane)
    //                   lanes16-31: elems 0-15 = K+16..31
    const int kb = k0 + lhalf * 16;
    v16bf bf[4];
    #pragma unroll
    for (int j = 0; j < 4; ++j) {
      if (PRECONV) bf[j] = *(const v16bf*)(wbrow[j] + kb);   // 32B bf16 vector load
      else         bf[j] = load_cvt_frag(wfrow[j] + kb, wfrow[j] + kb + 8);
    }

    if (k0 + 32 < HIDDEN) {
      __builtin_prefetch(xrow0 + ka + 32, 0, 3);   // global_prefetch_b8 (near)
      __builtin_prefetch(xrow1 + ka + 32, 0, 3);
    }

    #pragma unroll
    for (int j = 0; j < 4; ++j) {
      acc[0][j] = __builtin_amdgcn_wmma_f32_16x16x32_bf16(false, a0, false, bf[j],
                                                          (short)0, acc[0][j], false, false);
      acc[1][j] = __builtin_amdgcn_wmma_f32_16x16x32_bf16(false, a1, false, bf[j],
                                                          (short)0, acc[1][j], false, false);
    }
  }

  // C/D layout: VGPR r -> lanes 0-15: (M=r, N=lane); lanes 16-31: (M=r+8, N=lane-16)
  const int mbase = lhalf * 8;
  #pragma unroll
  for (int m = 0; m < 2; ++m)
    #pragma unroll
    for (int j = 0; j < 4; ++j)
      #pragma unroll
      for (int r = 0; r < 8; ++r)
        logits_lds[m * 16 + mbase + r][e_w + j * 16 + l15] = acc[m][j][r];

  __syncthreads();

  // ---- Phase 2: per-token sigmoid + bias + top-8 (one token per thread) ----
  if (threadIdx.x < M_TILE) {
    const int t = threadIdx.x;
    float top_sb[TOPK];   // selection key: sigmoid + bias (descending)
    float top_s [TOPK];   // raw sigmoid score (used for weights)
    int   top_i [TOPK];
    #pragma unroll
    for (int j = 0; j < TOPK; ++j) { top_sb[j] = -1e30f; top_s[j] = 0.f; top_i[j] = 0; }

    for (int e = 0; e < N_EXPERTS; ++e) {
      float logit = logits_lds[t][e];
      float s   = 1.0f / (1.0f + __expf(-logit));   // trans v_exp_f32
      float sb  = s + bias[e];
      int   idx = e;
      // Unrolled compare/swap insertion into a descending sorted list.
      // Strict '>' keeps lower indices first on ties, matching lax.top_k.
      #pragma unroll
      for (int j = 0; j < TOPK; ++j) {
        if (sb > top_sb[j]) {
          float tsb = top_sb[j]; top_sb[j] = sb;  sb  = tsb;
          float ts  = top_s[j];  top_s[j]  = s;   s   = ts;
          int   ti  = top_i[j];  top_i[j]  = idx; idx = ti;
        }
      }
    }

    float sum = 1e-20f;
    #pragma unroll
    for (int j = 0; j < TOPK; ++j) sum += top_s[j];
    const float scale = ROUTED_SCALING / sum;

    const size_t obase = (size_t)(token0 + t) * TOPK;
    #pragma unroll
    for (int j = 0; j < TOPK; ++j) {
      out_idx[obase + j] = top_i[j];
      out_w [obase + j]  = top_s[j] * scale;
    }
  }
}

extern "C" void kernel_launch(void* const* d_in, const int* in_sizes, int n_in,
                              void* d_out, int out_size, void* d_ws, size_t ws_size,
                              hipStream_t stream) {
  const float* x    = (const float*)d_in[0];
  const float* w    = (const float*)d_in[1];
  const float* bias = (const float*)d_in[2];
  // d_in[3] is top_k (== 8), fixed at compile time.

  const int tokens = in_sizes[0] / HIDDEN;          // 16384
  const int wElems = in_sizes[1];                   // 256 * 7168

  int*   out_idx = (int*)d_out;
  float* out_w   = (float*)d_out + (size_t)tokens * TOPK;

  dim3 block(128);                                  // 4 wave32s
  dim3 grid(tokens / M_TILE);                       // 512 blocks

  const size_t wBf16Bytes = (size_t)wElems * sizeof(__bf16);
  if (ws_size >= wBf16Bytes) {
    __bf16* wb = (__bf16*)d_ws;
    const int cThreads = wElems / 8;
    convert_w_bf16_kernel<<<(cThreads + 255) / 256, 256, 0, stream>>>(w, wb, wElems);
    moe_gate_topk_kernel<true><<<grid, block, 0, stream>>>(x, w, wb, bias, out_idx, out_w);
  } else {
    moe_gate_topk_kernel<false><<<grid, block, 0, stream>>>(x, w, (const __bf16*)nullptr,
                                                            bias, out_idx, out_w);
  }
}
